// SparseCrossAttention_65317862637920
// MI455X (gfx1250) — matrix operand order repaired and checked by reading, not compiled, not run on hardware
//
#include <hip/hip_runtime.h>
#include <hip/hip_bf16.h>

// ---------------------------------------------------------------------------
// SparseCrossAttention for MI455X (gfx1250, wave32, WMMA bf16 16x16x32)
//   B=2, Nq=Nk=2048, C=256, H=8, hd=32
// Memory-bound on similarities (268MB, streamed once, non-temporal). Fused
// attention: all 8 heads of a q-tile in one workgroup (cross-head mean via
// LDS, mask folded into the mean phase). Softmax runs in base-2 (log2e folded
// into the Q projection and the mean), exponentials are raw v_exp_f32, and
// row reductions are single-instruction DPP16 max/add (dpp applied to the
// max/add source itself: no mov_dpp, no canonicalization maxes).
// ---------------------------------------------------------------------------

typedef __attribute__((ext_vector_type(16))) __bf16 v16bf;
typedef __attribute__((ext_vector_type(8)))  __bf16 v8bf;
typedef __attribute__((ext_vector_type(8)))  float  v8f;

#define NB 2
#define NQ 2048
#define NK 2048
#define CDIM 256
#define NHEADS 8
#define HD 32
#define SCALE 0.17677669529663687f  // 1/sqrt(32)
#define LOG2E 1.4426950408889634f

static __device__ __forceinline__ v8f wmma_bf16(v16bf a, v16bf b, v8f c) {
  // D = A(16x32) * B(32x16) + C(16x16 f32)
  return __builtin_amdgcn_wmma_f32_16x16x32_bf16(false, a, false, b, (short)0, c,
                                                 false, false);
}

// ---- fused DPP16 reductions: one v_{max_num,add}_f32_dpp per step ----------
// Reduces over each 16-lane row of the wave32 (matches the WMMA C layout,
// where N = lane%16 and the two halves hold different M rows).
static __device__ __forceinline__ float red16_max(float v) {
  asm("v_max_num_f32_dpp %0, %0, %0 row_ror:8 row_mask:0xf bank_mask:0xf bound_ctrl:1" : "+v"(v));
  asm("v_max_num_f32_dpp %0, %0, %0 row_ror:4 row_mask:0xf bank_mask:0xf bound_ctrl:1" : "+v"(v));
  asm("v_max_num_f32_dpp %0, %0, %0 quad_perm:[2,3,0,1] row_mask:0xf bank_mask:0xf bound_ctrl:1" : "+v"(v));
  asm("v_max_num_f32_dpp %0, %0, %0 quad_perm:[1,0,3,2] row_mask:0xf bank_mask:0xf bound_ctrl:1" : "+v"(v));
  return v;
}
static __device__ __forceinline__ float red16_sum(float v) {
  asm("v_add_f32_dpp %0, %0, %0 row_ror:8 row_mask:0xf bank_mask:0xf bound_ctrl:1" : "+v"(v));
  asm("v_add_f32_dpp %0, %0, %0 row_ror:4 row_mask:0xf bank_mask:0xf bound_ctrl:1" : "+v"(v));
  asm("v_add_f32_dpp %0, %0, %0 quad_perm:[2,3,0,1] row_mask:0xf bank_mask:0xf bound_ctrl:1" : "+v"(v));
  asm("v_add_f32_dpp %0, %0, %0 quad_perm:[1,0,3,2] row_mask:0xf bank_mask:0xf bound_ctrl:1" : "+v"(v));
  return v;
}
// plain max without fcanonicalize insertion
static __device__ __forceinline__ float vmax(float a, float b) {
  float d;
  asm("v_max_num_f32 %0, %1, %2" : "=v"(d) : "v"(a), "v"(b));
  return d;
}

// --------------------------- weight prep -----------------------------------
// W (f32, [cin][cout]) -> Wt (bf16, [cout][cin]); 4 weights packed in dst.
__global__ __launch_bounds__(256) void sca_wprep(const float* __restrict__ w0,
                                                 const float* __restrict__ w1,
                                                 const float* __restrict__ w2,
                                                 const float* __restrict__ w3,
                                                 __bf16* __restrict__ dst) {
  int idx = blockIdx.x * 256 + threadIdx.x;     // 0 .. 4*65536-1
  int w = idx >> 16;
  int r = idx & 65535;
  int n = r >> 8;      // cout
  int k = r & 255;     // cin
  const float* s = (w == 0) ? w0 : (w == 1) ? w1 : (w == 2) ? w2 : w3;
  dst[(size_t)w * 65536 + (size_t)n * 256 + k] = (__bf16)s[(size_t)k * 256 + n];
}

// ------------------------------ GEMM ----------------------------------------
// out[M=4096, N=256] = (A[4096,256] * Bt^T) * ascale   (Bt: [cout][cin] bf16)
// MODE 0: A=f32, out=bf16 row-major (ascale folds softmax scale * log2e for Q)
// MODE 1: A=f32, out=bf16 transposed per (b,head): out[(b*256+n)*2048 + row%2048]
// MODE 2: A=bf16, out=f32 row-major + bias[n]
template <int MODE>
__global__ __launch_bounds__(256) void sca_gemm(const void* __restrict__ Ap,
                                                const __bf16* __restrict__ Bt,
                                                void* __restrict__ outp,
                                                const float* __restrict__ bias,
                                                float ascale) {
  const int lane = threadIdx.x & 31;
  const int wave = threadIdx.x >> 5;
  const int hf = lane >> 4;   // lane half (0/1)
  const int ln = lane & 15;
  const int tile = blockIdx.x * 8 + wave;   // 4096 tiles total
  const int rt = tile >> 4;                 // 0..255 (row tile)
  const int ct = tile & 15;                 // 0..15  (col tile)
  const int row = rt * 16 + ln;             // A row owned by this lane

  v8f acc = {0.f, 0.f, 0.f, 0.f, 0.f, 0.f, 0.f, 0.f};

#pragma unroll
  for (int k0 = 0; k0 < CDIM; k0 += 32) {
    v16bf a;
    if constexpr (MODE < 2) {
      const float* A = (const float*)Ap + (size_t)row * CDIM + k0;
      const float4 f0 = *(const float4*)(A + 16 * hf);
      const float4 f1 = *(const float4*)(A + 16 * hf + 4);
      const float4 f2 = *(const float4*)(A + 16 * hf + 8);
      const float4 f3 = *(const float4*)(A + 16 * hf + 12);
      a[0] = (__bf16)f0.x;  a[1] = (__bf16)f0.y;  a[2] = (__bf16)f0.z;  a[3] = (__bf16)f0.w;
      a[4] = (__bf16)f1.x;  a[5] = (__bf16)f1.y;  a[6] = (__bf16)f1.z;  a[7] = (__bf16)f1.w;
      a[8] = (__bf16)f2.x;  a[9] = (__bf16)f2.y;  a[10] = (__bf16)f2.z; a[11] = (__bf16)f2.w;
      a[12] = (__bf16)f3.x; a[13] = (__bf16)f3.y; a[14] = (__bf16)f3.z; a[15] = (__bf16)f3.w;
    } else {
      const __bf16* A = (const __bf16*)Ap + (size_t)row * CDIM + k0;
      v8bf lo = *(const v8bf*)(A + 16 * hf);
      v8bf hi = *(const v8bf*)(A + 16 * hf + 8);
      a = __builtin_shufflevector(lo, hi, 0, 1, 2, 3, 4, 5, 6, 7, 8, 9, 10, 11,
                                  12, 13, 14, 15);
    }
    // B: lane = column n, 16 contiguous K values starting at 16*half
    v16bf b = *(const v16bf*)(Bt + (size_t)(ct * 16 + ln) * CDIM + k0 + 16 * hf);
    acc = wmma_bf16(a, b, acc);
  }

#pragma unroll
  for (int r = 0; r < 8; ++r) {
    const int m = rt * 16 + r + 8 * hf;
    const int n = ct * 16 + ln;
    if constexpr (MODE == 0) {
      ((__bf16*)outp)[(size_t)m * CDIM + n] = (__bf16)(acc[r] * ascale);
    } else if constexpr (MODE == 1) {
      const int bb = m >> 11;          // batch
      const int nr = m & 2047;         // key position
      ((__bf16*)outp)[((size_t)(bb * 256 + n)) * NK + nr] = (__bf16)acc[r];
    } else {
      ((float*)outp)[(size_t)m * CDIM + n] = acc[r] + bias[n];
    }
  }
}

// --------------------------- fused attention --------------------------------
// grid: 256 blocks = (b, q-tile of 16); block: 256 threads = 8 waves = 8 heads
__global__ __launch_bounds__(256) void sca_attn(
    const __bf16* __restrict__ qb,   // [4096][256] bf16, pre-scaled by log2e/sqrt(hd)
    const __bf16* __restrict__ kb,   // [4096][256] bf16 row-major
    const __bf16* __restrict__ vtb,  // [(b*256 + h*32+d)][2048] bf16
    const float* __restrict__ sim,   // [2][8][2048][2048] f32
    const unsigned char* __restrict__ maskp,  // [2048][2048] bool
    __bf16* __restrict__ xb) {       // [4096][256] bf16 (attention output)
  __shared__ float s_sim[NHEADS][512];                  // 16 KB
  __shared__ float s_bias[512];                         //  2 KB (mean*log2e | +1e30)
  __shared__ __align__(16) __bf16 s_p[NHEADS][16][32];  //  8 KB

  const int b = blockIdx.x >> 7;
  const int q0 = (blockIdx.x & 127) << 4;
  const int h = threadIdx.x >> 5;   // wave == head
  const int lane = threadIdx.x & 31;
  const int hf = lane >> 4;
  const int ln = lane & 15;

  // Q tile (16x32) in A layout, loaded once (pre-scaled by log2e/sqrt(hd))
  const __bf16* qrow = qb + (size_t)(b * NQ + q0 + ln) * CDIM + h * HD;
  v8bf qlo = *(const v8bf*)(qrow + 16 * hf);
  v8bf qhi = *(const v8bf*)(qrow + 16 * hf + 8);
  v16bf aq = __builtin_shufflevector(qlo, qhi, 0, 1, 2, 3, 4, 5, 6, 7, 8, 9, 10,
                                     11, 12, 13, 14, 15);

  const float* simbase = sim + ((size_t)((b * NHEADS + h) * NQ + q0)) * NK;
  const __bf16* vbase = vtb + (size_t)(b * 256 + h * HD) * NK;

  // mean-phase geometry for this thread (positions p and p+256)
  // p = i*32 + lane32  ->  m = (i&7) + 8*(lane32>>4), col = 16*(i>>3) + (lane32&15)
  const int p0 = threadIdx.x;
  const int p1 = threadIdx.x + 256;
  const int m_p0 = (((p0 >> 5) & 7) + 8 * ((p0 >> 4) & 1));
  const int c_p0 = 16 * (p0 >> 8) + (p0 & 15);
  const int m_p1 = (((p1 >> 5) & 7) + 8 * ((p1 >> 4) & 1));
  const int c_p1 = 16 * (p1 >> 8) + (p1 & 15);
  const unsigned char* mrow0 = maskp + (size_t)(q0 + m_p0) * NK + c_p0;
  const unsigned char* mrow1 = maskp + (size_t)(q0 + m_p1) * NK + c_p1;

  float mrow[8], lrow[8];
  v8f acc0 = {0.f, 0.f, 0.f, 0.f, 0.f, 0.f, 0.f, 0.f};
  v8f acc1 = {0.f, 0.f, 0.f, 0.f, 0.f, 0.f, 0.f, 0.f};
#pragma unroll
  for (int r = 0; r < 8; ++r) { mrow[r] = -1e30f; lrow[r] = 0.f; }

  for (int kk = 0; kk < NK; kk += 32) {
    // ---- load own head's sim tile (16 rows x 32 cols), non-temporal ----
    float sv[16];
#pragma unroll
    for (int t = 0; t < 2; ++t)
#pragma unroll
      for (int r = 0; r < 8; ++r)
        sv[t * 8 + r] = __builtin_nontemporal_load(
            simbase + (size_t)(r + 8 * hf) * NK + kk + 16 * t + ln);
#pragma unroll
    for (int i = 0; i < 16; ++i) s_sim[h][i * 32 + lane] = sv[i];
    __syncthreads();

    // ---- block-wide cross-head mean (pre-scaled by log2e) fused with mask --
    {
      float s0 = 0.f, s1 = 0.f;
#pragma unroll
      for (int hh = 0; hh < NHEADS; ++hh) {
        s0 += s_sim[hh][p0];
        s1 += s_sim[hh][p1];
      }
      const unsigned char mk0 = mrow0[kk];
      const unsigned char mk1 = mrow1[kk];
      s_bias[p0] = mk0 ? s0 * (0.125f * LOG2E) : 1e30f;
      s_bias[p1] = mk1 ? s1 * (0.125f * LOG2E) : 1e30f;
    }
    __syncthreads();

    // ---- C operand of QK^T WMMA carries log2e*(sim - mean) | -1e30 ----
    v8f c0, c1;
#pragma unroll
    for (int r = 0; r < 8; ++r) {
      c0[r] = fmaf(sv[r], LOG2E, -s_bias[r * 32 + lane]);
      c1[r] = fmaf(sv[8 + r], LOG2E, -s_bias[(8 + r) * 32 + lane]);
    }

    // ---- S2 = log2e*(Q*K^T/sqrt(hd) + sim - mean)  (base-2 scores) ----
    v16bf bk0 = *(const v16bf*)(kb + (size_t)(b * NK + kk + ln) * CDIM + h * HD + 16 * hf);
    v16bf bk1 = *(const v16bf*)(kb + (size_t)(b * NK + kk + 16 + ln) * CDIM + h * HD + 16 * hf);
    v8f s0 = wmma_bf16(aq, bk0, c0);
    v8f s1 = wmma_bf16(aq, bk1, c1);

    // ---- online softmax (base 2): fused-DPP row reductions ----
    float pv[16];
#pragma unroll
    for (int r = 0; r < 8; ++r) {
      const float v = red16_max(vmax(s0[r], s1[r]));
      const float mn = vmax(mrow[r], v);
      const float alpha = __builtin_amdgcn_exp2f(mrow[r] - mn);
      mrow[r] = mn;
      pv[r] = __builtin_amdgcn_exp2f(s0[r] - mn);
      pv[8 + r] = __builtin_amdgcn_exp2f(s1[r] - mn);
      const float rs = red16_sum(pv[r] + pv[8 + r]);
      lrow[r] = lrow[r] * alpha + rs;
      acc0[r] *= alpha;
      acc1[r] *= alpha;
    }

    // ---- P (C layout) -> A layout via per-wave LDS bounce ----
#pragma unroll
    for (int t = 0; t < 2; ++t)
#pragma unroll
      for (int r = 0; r < 8; ++r)
        s_p[h][r + 8 * hf][16 * t + ln] = (__bf16)pv[t * 8 + r];
    v8bf plo = *(const v8bf*)&s_p[h][ln][16 * hf];
    v8bf phi = *(const v8bf*)&s_p[h][ln][16 * hf + 8];
    v16bf ap = __builtin_shufflevector(plo, phi, 0, 1, 2, 3, 4, 5, 6, 7, 8, 9,
                                       10, 11, 12, 13, 14, 15);

    // ---- O += P * V  (V transposed: column = head dim, contiguous in k) ----
    v16bf bv0 = *(const v16bf*)(vbase + (size_t)ln * NK + kk + 16 * hf);
    v16bf bv1 = *(const v16bf*)(vbase + (size_t)(16 + ln) * NK + kk + 16 * hf);
    acc0 = wmma_bf16(ap, bv0, acc0);
    acc1 = wmma_bf16(ap, bv1, acc1);
    // no end-of-loop barrier needed: next iteration's s_sim writes are fenced
    // by this iteration's barriers; s_bias is rewritten only after barrier 1.
  }

  // ---- normalize and write x[b, q, h*32+d] ----
#pragma unroll
  for (int r = 0; r < 8; ++r) {
    const float inv = 1.0f / lrow[r];
    const int m = r + 8 * hf;
    size_t base = (size_t)(b * NQ + q0 + m) * CDIM + h * HD;
    xb[base + ln] = (__bf16)(acc0[r] * inv);
    xb[base + 16 + ln] = (__bf16)(acc1[r] * inv);
  }
}

// ----------------------------- launch ---------------------------------------
extern "C" void kernel_launch(void* const* d_in, const int* in_sizes, int n_in,
                              void* d_out, int out_size, void* d_ws,
                              size_t ws_size, hipStream_t stream) {
  (void)in_sizes; (void)n_in; (void)out_size; (void)ws_size;
  const float* query = (const float*)d_in[0];
  const float* key   = (const float*)d_in[1];
  const float* value = (const float*)d_in[2];
  // d_in[3] qpos, d_in[4] kpos unused (rope is None in reference)
  const unsigned char* mask = (const unsigned char*)d_in[5];
  const float* sim = (const float*)d_in[6];
  const float* Wq = (const float*)d_in[7];
  const float* Wk = (const float*)d_in[8];
  const float* Wv = (const float*)d_in[9];
  const float* Wp = (const float*)d_in[10];
  const float* bp = (const float*)d_in[11];

  char* ws = (char*)d_ws;
  __bf16* qb  = (__bf16*)(ws + (size_t)0);
  __bf16* kb  = (__bf16*)(ws + ((size_t)2 << 20));
  __bf16* vtb = (__bf16*)(ws + ((size_t)4 << 20));
  __bf16* xb  = (__bf16*)(ws + ((size_t)6 << 20));
  __bf16* wt  = (__bf16*)(ws + ((size_t)8 << 20));  // 4 x [256][256] bf16

  sca_wprep<<<1024, 256, 0, stream>>>(Wq, Wk, Wv, Wp, wt);
  sca_gemm<0><<<512, 256, 0, stream>>>(query, wt, qb, nullptr, SCALE * LOG2E);
  sca_gemm<0><<<512, 256, 0, stream>>>(key, wt + 65536, kb, nullptr, 1.0f);
  sca_gemm<1><<<512, 256, 0, stream>>>(value, wt + 2 * 65536, vtb, nullptr, 1.0f);
  sca_attn<<<256, 256, 0, stream>>>(qb, kb, vtb, sim, mask, xb);
  sca_gemm<2><<<512, 256, 0, stream>>>(xb, wt + 3 * 65536, d_out, bp, 1.0f);
}
